// MoE_56118042690096
// MI455X (gfx1250) — matrix operand order repaired and checked by reading, not compiled, not run on hardware
//
#include <hip/hip_runtime.h>

// ---------------- model constants (match reference) ----------------
#define BB   2
#define SS   1024
#define DD   1024
#define HH   2048
#define QHN  16
#define KHN  8
#define EE   5
#define LL   2
#define VV   32000
#define DHD  64
#define TT   (BB*SS)   // 2048 tokens

// GEMM tiling (all problem dims are multiples of 64, K multiples of 32)
#define BM   64
#define BN   64
#define BK   32

// ---------------- WMMA types ----------------
typedef __attribute__((ext_vector_type(16))) __bf16 v16bf;
typedef __attribute__((ext_vector_type(8)))  float  v8f;

union Frag16 { v16bf v; unsigned short s[16]; };

static __device__ __forceinline__ unsigned int f2bf_pair(float lo, float hi) {
    unsigned int ul = __float_as_uint(lo);
    ul += 0x7fffu + ((ul >> 16) & 1u);          // RNE
    unsigned int uh = __float_as_uint(hi);
    uh += 0x7fffu + ((uh >> 16) & 1u);
    return (ul >> 16) | (uh & 0xffff0000u);
}

// =====================================================================
// GEMM: C[M,N] (+)= rowScale[m] * (A[M,K] @ W[N,K]^T + bias[n])
// bf16 WMMA 16x16x32, fp32 accumulate. 128 threads = 4 waves.
// Block tile 64x64; waves in 2x2 grid; each wave owns a 32x32 sub-tile
// (2 A-frags x 2 B-frags -> 4 WMMAs per K-step, quadratic reuse).
// Dims must satisfy M%64==0, N%64==0, K%32==0 (true for this model).
// =====================================================================
__global__ __launch_bounds__(128)
void gemm_bf16_wmma(const float* __restrict__ A, const float* __restrict__ W,
                    const float* __restrict__ bias, float* __restrict__ C,
                    int M, int N, int Kd,
                    const float* __restrict__ rowScale, int rsStride,
                    int accumulate)
{
    __shared__ unsigned short As[BM][BK + 2];
    __shared__ unsigned short Bs[BN][BK + 2];

    const int tid  = threadIdx.x;
    const int lane = tid & 31;
    const int wave = tid >> 5;
    const int row0 = blockIdx.y * BM;
    const int col0 = blockIdx.x * BN;
    const int wr = (wave >> 1) * 32;   // 0 or 32
    const int wc = (wave & 1) * 32;    // 0 or 32

    v8f acc00 = {0.f,0.f,0.f,0.f,0.f,0.f,0.f,0.f};
    v8f acc01 = acc00, acc10 = acc00, acc11 = acc00;

    for (int k0 = 0; k0 < Kd; k0 += BK) {
        // ---- stage fp32 -> bf16 tiles (b128 loads, b32 LDS stores).
        // 64x32 floats per matrix = 512 float4; 128 threads -> 4 fixed iters.
        #pragma unroll
        for (int it = 0; it < 4; ++it) {
            const int i  = it * 128 + tid;
            const int r  = i >> 3;          // 8 float4 per row
            const int c4 = (i & 7) * 4;
            const float* ap = A + (size_t)(row0 + r) * Kd + k0 + c4;
            const float* wp = W + (size_t)(col0 + r) * Kd + k0 + c4;
            const float4 va = *(const float4*)ap;
            *(unsigned int*)&As[r][c4]     = f2bf_pair(va.x, va.y);
            *(unsigned int*)&As[r][c4 + 2] = f2bf_pair(va.z, va.w);
            const float4 vw = *(const float4*)wp;
            *(unsigned int*)&Bs[r][c4]     = f2bf_pair(vw.x, vw.y);
            *(unsigned int*)&Bs[r][c4 + 2] = f2bf_pair(vw.z, vw.w);
            // speculative prefetch of next K tile (dropped if invalid)
            __builtin_prefetch(ap + BK, 0, 1);
            __builtin_prefetch(wp + BK, 0, 1);
        }
        __syncthreads();

        // ---- build fragments per documented CDNA5 16-bit layouts
        Frag16 fa0, fa1, fb0, fb1;
        const int arow = wr + (lane & 15);
        const int bn   = wc + (lane & 15);
        const int hi   = lane >> 4;
        #pragma unroll
        for (int vv = 0; vv < 8; ++vv) {
            // A 16x32: lanes0-15 K={0..7,16..23}, lanes16-31 K={8..15,24..31}
            const int klo = ((vv & 3) << 1) + ((vv >> 2) << 4) + (hi << 3);
            fa0.s[2 * vv]     = As[arow][klo];
            fa0.s[2 * vv + 1] = As[arow][klo + 1];
            fa1.s[2 * vv]     = As[arow + 16][klo];
            fa1.s[2 * vv + 1] = As[arow + 16][klo + 1];
            // B 32x16: lanes0-15 K={0..7,16..23}, lanes16-31 K={8..15,24..31}; N = lane&15
            const int bk = ((vv & 3) << 1) + ((vv >> 2) << 4) + (hi << 3);
            fb0.s[2 * vv]     = Bs[bn][bk];
            fb0.s[2 * vv + 1] = Bs[bn][bk + 1];
            fb1.s[2 * vv]     = Bs[bn + 16][bk];
            fb1.s[2 * vv + 1] = Bs[bn + 16][bk + 1];
        }
        acc00 = __builtin_amdgcn_wmma_f32_16x16x32_bf16(
            false, fa0.v, false, fb0.v, (short)0, acc00, false, false);
        acc01 = __builtin_amdgcn_wmma_f32_16x16x32_bf16(
            false, fa0.v, false, fb1.v, (short)0, acc01, false, false);
        acc10 = __builtin_amdgcn_wmma_f32_16x16x32_bf16(
            false, fa1.v, false, fb0.v, (short)0, acc10, false, false);
        acc11 = __builtin_amdgcn_wmma_f32_16x16x32_bf16(
            false, fa1.v, false, fb1.v, (short)0, acc11, false, false);
        __syncthreads();
    }

    // ---- store: f32 C/D 16x16: VGPR v -> M = v + 8*(lane>=16), N = lane&15
    #pragma unroll
    for (int vv = 0; vv < 8; ++vv) {
        const int m0 = row0 + wr + vv + ((lane >> 4) << 3);
        const int n0 = col0 + wc + (lane & 15);
        const float b0 = bias ? bias[n0]      : 0.f;
        const float b1v = bias ? bias[n0 + 16] : 0.f;
        {
            const float scale = rowScale ? rowScale[(size_t)m0 * rsStride] : 1.f;
            const size_t idx = (size_t)m0 * N + n0;
            C[idx]      = (accumulate ? C[idx]      : 0.f) + scale * (acc00[vv] + b0);
            C[idx + 16] = (accumulate ? C[idx + 16] : 0.f) + scale * (acc01[vv] + b1v);
        }
        {
            const int m1 = m0 + 16;
            const float scale = rowScale ? rowScale[(size_t)m1 * rsStride] : 1.f;
            const size_t idx = (size_t)m1 * N + n0;
            C[idx]      = (accumulate ? C[idx]      : 0.f) + scale * (acc10[vv] + b0);
            C[idx + 16] = (accumulate ? C[idx + 16] : 0.f) + scale * (acc11[vv] + b1v);
        }
    }
}

// =====================================================================
// Embedding gather
// =====================================================================
__global__ void embed_kernel(const int* __restrict__ x,
                             const float* __restrict__ emb,
                             float* __restrict__ h)
{
    int t = blockIdx.x;
    int tok = x[t];
    for (int d = threadIdx.x; d < DD; d += blockDim.x)
        h[(size_t)t * DD + d] = emb[(size_t)tok * DD + d];
}

// =====================================================================
// RMSNorm: y = g * x * rsqrt(mean(x^2) + 1e-5), one block per token
// =====================================================================
__global__ __launch_bounds__(256)
void rmsnorm_kernel(const float* __restrict__ x, const float* __restrict__ g,
                    float* __restrict__ y)
{
    __shared__ float red[256];
    int t = blockIdx.x;
    const float* xr = x + (size_t)t * DD;
    float s = 0.f;
    for (int d = threadIdx.x; d < DD; d += 256) { float v = xr[d]; s += v * v; }
    red[threadIdx.x] = s;
    __syncthreads();
    for (int off = 128; off > 0; off >>= 1) {
        if (threadIdx.x < off) red[threadIdx.x] += red[threadIdx.x + off];
        __syncthreads();
    }
    float inv = rsqrtf(red[0] / (float)DD + 1e-5f);
    for (int d = threadIdx.x; d < DD; d += 256)
        y[(size_t)t * DD + d] = g[d] * xr[d] * inv;
}

// =====================================================================
// RoPE in-place over (T, NH, 64); one thread per (even,odd) pair
// =====================================================================
__global__ void rope_kernel(float* __restrict__ buf, int NH,
                            const int* __restrict__ startposPtr, int totalPairs)
{
    int i = blockIdx.x * blockDim.x + threadIdx.x;
    if (i >= totalPairs) return;
    int p    = i & 31;          // DH/2 = 32 pairs
    int rest = i >> 5;
    int t    = rest / NH;
    int spos = t % SS;
    float freq = powf(50000.0f, -(float)(2 * p) / 64.0f);
    float ang  = (float)(startposPtr[0] + spos) * freq;
    float c = cosf(ang), sn = sinf(ang);
    size_t base = (size_t)i * 2;
    float x0 = buf[base], x1 = buf[base + 1];
    buf[base]     = x0 * c - x1 * sn;
    buf[base + 1] = x0 * sn + x1 * c;
}

// =====================================================================
// Causal attention with GQA (SCALE=2). One block per (b, h, q).
// =====================================================================
__global__ __launch_bounds__(128)
void attn_kernel(const float* __restrict__ q, const float* __restrict__ k,
                 const float* __restrict__ v, float* __restrict__ o)
{
    __shared__ float qs[DHD];
    __shared__ float sc[SS];
    __shared__ float red[128];

    const int tid = threadIdx.x;
    const int qi = blockIdx.x, h = blockIdx.y, b = blockIdx.z;
    const int kh = h >> 1;          // SCALE = QH/KH = 2
    const int nk = qi + 1;

    const float* qp = q + (((size_t)(b * SS + qi)) * QHN + h) * DHD;
    if (tid < DHD) qs[tid] = qp[tid];
    __syncthreads();

    float mx = -3.4e38f;
    for (int j = tid; j < nk; j += 128) {
        const float* kp = k + (((size_t)(b * SS + j)) * KHN + kh) * DHD;
        float dot = 0.f;
        #pragma unroll 8
        for (int d = 0; d < DHD; ++d) dot += qs[d] * kp[d];
        dot *= 0.125f;              // 1/sqrt(64)
        sc[j] = dot;
        mx = fmaxf(mx, dot);
    }
    red[tid] = mx; __syncthreads();
    for (int off = 64; off > 0; off >>= 1) {
        if (tid < off) red[tid] = fmaxf(red[tid], red[tid + off]);
        __syncthreads();
    }
    mx = red[0]; __syncthreads();

    float ls = 0.f;
    for (int j = tid; j < nk; j += 128) {
        float e = expf(sc[j] - mx);
        sc[j] = e; ls += e;
    }
    red[tid] = ls; __syncthreads();
    for (int off = 64; off > 0; off >>= 1) {
        if (tid < off) red[tid] += red[tid + off];
        __syncthreads();
    }
    float inv = 1.f / red[0];

    if (tid < DHD) {
        float accv = 0.f;
        for (int j = 0; j < nk; ++j)
            accv += sc[j] * v[(((size_t)(b * SS + j)) * KHN + kh) * DHD + tid];
        o[(((size_t)(b * SS + qi)) * QHN + h) * DHD + tid] = accv * inv;
    }
}

// =====================================================================
// Gating: logits = xn @ Wg^T (E=5), top-2 softmax -> cw[T,E]
// One wave (32 lanes) per token, 4 tokens per 128-thread block.
// =====================================================================
__global__ __launch_bounds__(128)
void gate_kernel(const float* __restrict__ xn, const float* __restrict__ Wg,
                 float* __restrict__ cw)
{
    int wave = threadIdx.x >> 5;
    int lane = threadIdx.x & 31;
    int t = blockIdx.x * 4 + wave;
    if (t >= TT) return;

    const float* xr = xn + (size_t)t * DD;
    float logit[EE];
    #pragma unroll
    for (int e = 0; e < EE; ++e) {
        const float* w = Wg + (size_t)e * DD;
        float s = 0.f;
        for (int d = lane; d < DD; d += 32) s += xr[d] * w[d];
        #pragma unroll
        for (int off = 16; off > 0; off >>= 1) s += __shfl_xor(s, off, 32);
        logit[e] = s;
    }
    if (lane == 0) {
        int i0 = 0;
        for (int e = 1; e < EE; ++e) if (logit[e] > logit[i0]) i0 = e;
        int i1 = -1;
        for (int e = 0; e < EE; ++e) {
            if (e == i0) continue;
            if (i1 < 0 || logit[e] > logit[i1]) i1 = e;
        }
        float e1 = expf(logit[i1] - logit[i0]);
        float inv = 1.f / (1.f + e1);
        #pragma unroll
        for (int e = 0; e < EE; ++e) cw[(size_t)t * EE + e] = 0.f;
        cw[(size_t)t * EE + i0] = inv;
        cw[(size_t)t * EE + i1] = e1 * inv;
    }
}

// =====================================================================
// g = h1 * silu(h2) in-place into h1
// =====================================================================
__global__ void silu_mul_kernel(float* __restrict__ h1,
                                const float* __restrict__ h2, size_t n)
{
    size_t i = (size_t)blockIdx.x * blockDim.x + threadIdx.x;
    if (i < n) {
        float a = h1[i], b = h2[i];
        h1[i] = a * (b / (1.f + expf(-b)));
    }
}

// =====================================================================
// Orchestration
// =====================================================================
extern "C" void kernel_launch(void* const* d_in, const int* in_sizes, int n_in,
                              void* d_out, int out_size, void* d_ws, size_t ws_size,
                              hipStream_t stream)
{
    (void)in_sizes; (void)n_in; (void)out_size; (void)ws_size;

    const int*   x          = (const int*)  d_in[0];
    const int*   startpos   = (const int*)  d_in[1];
    const float* emb        = (const float*)d_in[2];
    const float* attn_scale = (const float*)d_in[3];
    const float* Wq         = (const float*)d_in[4];
    const float* bq         = (const float*)d_in[5];
    const float* Wk         = (const float*)d_in[6];
    const float* bk         = (const float*)d_in[7];
    const float* Wv         = (const float*)d_in[8];
    const float* bv         = (const float*)d_in[9];
    const float* Wo         = (const float*)d_in[10];
    const float* bo         = (const float*)d_in[11];
    const float* exp_scale  = (const float*)d_in[12];
    const float* Wgate      = (const float*)d_in[13];
    const float* W1         = (const float*)d_in[14];
    const float* b1         = (const float*)d_in[15];
    const float* W2         = (const float*)d_in[16];
    const float* b2         = (const float*)d_in[17];
    const float* W3         = (const float*)d_in[18];
    const float* b3         = (const float*)d_in[19];
    const float* fin_scale  = (const float*)d_in[20];
    float* out = (float*)d_out;

    // workspace layout (floats)
    float* ws = (float*)d_ws;
    float* h  = ws;                              // T*D
    float* xn = h  + (size_t)TT * DD;            // T*D
    float* qb = xn + (size_t)TT * DD;            // T*QH*DH = T*D
    float* kb = qb + (size_t)TT * DD;            // T*KH*DH
    float* vb = kb + (size_t)TT * KHN * DHD;     // T*KH*DH
    float* ob = vb + (size_t)TT * KHN * DHD;     // T*D
    float* h1 = ob + (size_t)TT * DD;            // T*H
    float* h2 = h1 + (size_t)TT * HH;            // T*H
    float* cw = h2 + (size_t)TT * HH;            // T*E

    const dim3 gblk(128);
    const dim3 gq(DD / BN,            TT / BM);  // N=1024
    const dim3 gkv((KHN * DHD) / BN,  TT / BM);  // N=512
    const dim3 gh(HH / BN,            TT / BM);  // N=2048
    const dim3 gd(DD / BN,            TT / BM);  // N=1024
    const dim3 gv(VV / BN,            TT / BM);  // N=32000

    embed_kernel<<<TT, 256, 0, stream>>>(x, emb, h);

    for (int l = 0; l < LL; ++l) {
        // ---- attention block ----
        rmsnorm_kernel<<<TT, 256, 0, stream>>>(h, attn_scale + (size_t)l * DD, xn);

        gemm_bf16_wmma<<<gq, gblk, 0, stream>>>(xn, Wq + (size_t)l * DD * DD,
            bq + (size_t)l * DD, qb, TT, DD, DD, nullptr, 0, 0);
        gemm_bf16_wmma<<<gkv, gblk, 0, stream>>>(xn, Wk + (size_t)l * KHN * DHD * DD,
            bk + (size_t)l * KHN * DHD, kb, TT, KHN * DHD, DD, nullptr, 0, 0);
        gemm_bf16_wmma<<<gkv, gblk, 0, stream>>>(xn, Wv + (size_t)l * KHN * DHD * DD,
            bv + (size_t)l * KHN * DHD, vb, TT, KHN * DHD, DD, nullptr, 0, 0);

        const int qpairs = TT * QHN * (DHD / 2);
        const int kpairs = TT * KHN * (DHD / 2);
        rope_kernel<<<(qpairs + 255) / 256, 256, 0, stream>>>(qb, QHN, startpos, qpairs);
        rope_kernel<<<(kpairs + 255) / 256, 256, 0, stream>>>(kb, KHN, startpos, kpairs);

        attn_kernel<<<dim3(SS, QHN, BB), 128, 0, stream>>>(qb, kb, vb, ob);

        // h += o @ Wo^T + bo
        gemm_bf16_wmma<<<gd, gblk, 0, stream>>>(ob, Wo + (size_t)l * DD * DD,
            bo + (size_t)l * DD, h, TT, DD, DD, nullptr, 0, 1);

        // ---- MoE block ----
        rmsnorm_kernel<<<TT, 256, 0, stream>>>(h, exp_scale + (size_t)l * DD, xn);
        gate_kernel<<<(TT + 3) / 4, 128, 0, stream>>>(xn, Wgate + (size_t)l * EE * DD, cw);

        for (int e = 0; e < EE; ++e) {
            const float* W1e = W1 + ((size_t)l * EE + e) * HH * DD;
            const float* b1e = b1 + ((size_t)l * EE + e) * HH;
            const float* W2e = W2 + ((size_t)l * EE + e) * HH * DD;
            const float* b2e = b2 + ((size_t)l * EE + e) * HH;
            const float* W3e = W3 + ((size_t)l * EE + e) * DD * HH;
            const float* b3e = b3 + ((size_t)l * EE + e) * DD;

            gemm_bf16_wmma<<<gh, gblk, 0, stream>>>(xn, W1e, b1e, h1,
                TT, HH, DD, nullptr, 0, 0);
            gemm_bf16_wmma<<<gh, gblk, 0, stream>>>(xn, W2e, b2e, h2,
                TT, HH, DD, nullptr, 0, 0);

            const size_t nelem = (size_t)TT * HH;
            silu_mul_kernel<<<(unsigned)((nelem + 255) / 256), 256, 0, stream>>>(h1, h2, nelem);

            // h += cw[:,e] * (g @ W3^T + b3)
            gemm_bf16_wmma<<<gd, gblk, 0, stream>>>(h1, W3e, b3e, h,
                TT, DD, HH, cw + e, EE, 1);
        }
    }

    // final norm + logits
    rmsnorm_kernel<<<TT, 256, 0, stream>>>(h, fin_scale, xn);
    gemm_bf16_wmma<<<gv, gblk, 0, stream>>>(xn, emb, nullptr, out,
        TT, VV, DD, nullptr, 0, 0);
}
